// TextLSTM_3942779977872
// MI455X (gfx1250) — matrix operand order repaired and checked by reading, not compiled, not run on hardware
//
#include <hip/hip_runtime.h>
#include <hip/hip_bf16.h>

// ---------------------------------------------------------------------------
// MI455X (gfx1250) LSTM-LM forward.
//  - bf16 WMMA (v_wmma_f32_16x16x32_bf16) for all matmuls, f32 accumulate
//  - persistent per-layer recurrent kernel with software grid barrier
//  - weights repacked once per call to [N][K] bf16 for contiguous B-frag loads
// ---------------------------------------------------------------------------

typedef __bf16 bf16;
typedef bf16  v8bf  __attribute__((ext_vector_type(8)));
typedef bf16  v16bf __attribute__((ext_vector_type(16)));
typedef float v8f   __attribute__((ext_vector_type(8)));

// ---- WMMA fragment helpers (layouts per CDNA5 ISA §7.12.2) -----

// A-matrix 16x32 bf16: lane L: m = L&15, khalf = L>>4.
// vector elems [0..7]  = A[m][khalf*8 + 0..7]
// vector elems [8..15] = A[m][16 + khalf*8 + 0..7]
__device__ inline v16bf load_a_frag(const bf16* __restrict__ base, int stride, int lane) {
    const int m = lane & 15, kh = lane >> 4;
    const bf16* p = base + (size_t)m * stride + kh * 8;
    v8bf lo = *(const v8bf*)p;
    v8bf hi = *(const v8bf*)(p + 16);
    v16bf r;
#pragma unroll
    for (int i = 0; i < 8; ++i) { r[i] = lo[i]; r[i + 8] = hi[i]; }
    return r;
}

// B-matrix 32x16 bf16 from transposed storage Bt[n][k]:
// lane L: n = L&15, khalf = L>>4; elems [0..7] = B[khalf*8 + 0..7][n], elems [8..15] = B[16 + khalf*8 + 0..7][n]
__device__ inline v16bf load_b_frag(const bf16* __restrict__ baseT, int stride, int lane) {
    const int n = lane & 15, kh = lane >> 4;
    const bf16* p = baseT + (size_t)n * stride + kh * 8;
    v8bf lo = *(const v8bf*)p;
    v8bf hi = *(const v8bf*)(p + 16);
    v16bf r;
#pragma unroll
    for (int i = 0; i < 8; ++i) { r[i] = lo[i]; r[i + 8] = hi[i]; }
    return r;
}

__device__ inline v8f wmma_bf16(v16bf a, v16bf b, v8f c) {
    // (neg_a, A, neg_b, B, c_mod, C, reuse_a, reuse_b)
    return __builtin_amdgcn_wmma_f32_16x16x32_bf16(false, a, false, b, (short)0, c,
                                                   false, false);
}

__device__ inline float sigmoid_(float x) { return 1.0f / (1.0f + __expf(-x)); }
__device__ inline float tanh_(float x) {
    float ax = fabsf(x);
    float e  = __expf(-2.0f * ax);
    float t  = (1.0f - e) / (1.0f + e);
    return copysignf(t, x);
}

// ---------------------------------------------------------------------------
// Transpose + f32->bf16 convert:  in[z][K][N] (f32)  ->  out[z][N][K] (bf16)
// ---------------------------------------------------------------------------
__global__ void __launch_bounds__(256)
transpose_f32_to_bf16T(const float* __restrict__ in, bf16* __restrict__ outT,
                       int K, int N) {
    __shared__ float tile[32][33];
    const size_t slice = (size_t)K * N * blockIdx.z;
    const float* inp = in + slice;
    bf16* outp = outT + slice;
    const int k0 = blockIdx.x * 32, n0 = blockIdx.y * 32;
    const int tx = threadIdx.x & 31, ty = threadIdx.x >> 5;
#pragma unroll
    for (int r = ty; r < 32; r += 8)
        tile[r][tx] = inp[(size_t)(k0 + r) * N + (n0 + tx)];
    __syncthreads();
#pragma unroll
    for (int r = ty; r < 32; r += 8)
        outp[(size_t)(n0 + r) * K + (k0 + tx)] = (bf16)tile[tx][r];
}

// ---------------------------------------------------------------------------
// Embedding gather:  seq[(s*64+b)][e] = (bf16) emb[X[b][s]][e]
// ---------------------------------------------------------------------------
__global__ void __launch_bounds__(256)
embed_kernel(const int* __restrict__ X, const float* __restrict__ emb,
             bf16* __restrict__ seq) {
    const int sb = blockIdx.x;                 // 0 .. 16383, sb = s*64 + b
    const int s = sb >> 6, b = sb & 63;
    const int tok = X[b * 256 + s];
    const float* src = emb + (size_t)tok * 1024;
    bf16* dst = seq + (size_t)sb * 1024;
    for (int e = threadIdx.x; e < 1024; e += 256) dst[e] = (bf16)src[e];
}

// ---------------------------------------------------------------------------
// Tiled WMMA GEMM with bias:  C[M][N] (f32) = A[M][K] (bf16) x Bt^T + bias[N]
//   Bt is stored [N][K].  Macro tile 64(M) x 128(N), 8 waves, 2x2 frags/wave.
//   Requires M%64==0, N%128==0, K%32==0.
// ---------------------------------------------------------------------------
__global__ void __launch_bounds__(256)
wmma_gemm_bias_kernel(const bf16* __restrict__ A, const bf16* __restrict__ Bt,
                      const float* __restrict__ bias, float* __restrict__ C,
                      int M, int N, int K) {
    __shared__ __align__(16) bf16 ldsA[64 * 40];    // 32 k + 8 pad per row
    __shared__ __align__(16) bf16 ldsB[128 * 40];

    const int tid  = threadIdx.x;
    const int lane = tid & 31;
    const int wid  = tid >> 5;       // 0..7
    const int wm   = wid >> 2;       // 0..1 -> 32 rows
    const int wn   = wid & 3;        // 0..3 -> 32 cols
    const int n0   = blockIdx.x * 128;
    const int m0   = blockIdx.y * 64;

    v8f acc[2][2];
#pragma unroll
    for (int i = 0; i < 2; ++i)
#pragma unroll
        for (int j = 0; j < 2; ++j)
#pragma unroll
            for (int e = 0; e < 8; ++e) acc[i][j][e] = 0.0f;

    const int arow = tid >> 2, aseg = tid & 3;   // 64 rows x 4 segs of 8 elems
    const int brow = tid >> 1, bseg = tid & 1;   // 128 rows x 2 segs of 16 elems

    for (int kc = 0; kc < K; kc += 32) {
        v8bf av  = *(const v8bf*)(A  + (size_t)(m0 + arow) * K + kc + aseg * 8);
        v8bf bv0 = *(const v8bf*)(Bt + (size_t)(n0 + brow) * K + kc + bseg * 16);
        v8bf bv1 = *(const v8bf*)(Bt + (size_t)(n0 + brow) * K + kc + bseg * 16 + 8);
        if (kc + 32 < K) {  // global_prefetch_b8 for the next k-chunk
            __builtin_prefetch(A  + (size_t)(m0 + arow) * K + kc + 32, 0, 1);
            __builtin_prefetch(Bt + (size_t)(n0 + brow) * K + kc + 32, 0, 1);
        }
        __syncthreads();   // protect previous iteration's LDS reads
        *(v8bf*)(ldsA + arow * 40 + aseg * 8)       = av;
        *(v8bf*)(ldsB + brow * 40 + bseg * 16)      = bv0;
        *(v8bf*)(ldsB + brow * 40 + bseg * 16 + 8)  = bv1;
        __syncthreads();

        v16bf af0 = load_a_frag(ldsA + (wm * 32 + 0)  * 40, 40, lane);
        v16bf af1 = load_a_frag(ldsA + (wm * 32 + 16) * 40, 40, lane);
        v16bf bf0 = load_b_frag(ldsB + (wn * 32 + 0)  * 40, 40, lane);
        v16bf bf1 = load_b_frag(ldsB + (wn * 32 + 16) * 40, 40, lane);

        acc[0][0] = wmma_bf16(af0, bf0, acc[0][0]);
        acc[0][1] = wmma_bf16(af0, bf1, acc[0][1]);
        acc[1][0] = wmma_bf16(af1, bf0, acc[1][0]);
        acc[1][1] = wmma_bf16(af1, bf1, acc[1][1]);
    }

    // Epilogue: D layout — lane L: n = L&15; VGPR j -> m = j + 8*(L>>4)
    const int cn = lane & 15, ckh = lane >> 4;
#pragma unroll
    for (int am = 0; am < 2; ++am)
#pragma unroll
        for (int an = 0; an < 2; ++an) {
            const int col = n0 + wn * 32 + an * 16 + cn;
            const float bv = bias ? bias[col] : 0.0f;
#pragma unroll
            for (int j = 0; j < 8; ++j) {
                const int row = m0 + wm * 32 + am * 16 + j + 8 * ckh;
                C[(size_t)row * N + col] = acc[am][an][j] + bv;
            }
        }
}

// ---------------------------------------------------------------------------
// Software grid barrier (persistent recurrent kernel)
// ---------------------------------------------------------------------------
__device__ inline void grid_barrier(unsigned* counter, unsigned* gen, unsigned nblocks) {
    __syncthreads();
    if (threadIdx.x == 0) {
        __threadfence();  // make hseq/cstate stores device-visible
        unsigned g = __hip_atomic_load(gen, __ATOMIC_RELAXED, __HIP_MEMORY_SCOPE_AGENT);
        unsigned arrived =
            __hip_atomic_fetch_add(counter, 1u, __ATOMIC_ACQ_REL, __HIP_MEMORY_SCOPE_AGENT);
        if (arrived == nblocks - 1) {
            __hip_atomic_store(counter, 0u, __ATOMIC_RELAXED, __HIP_MEMORY_SCOPE_AGENT);
            __hip_atomic_store(gen, g + 1u, __ATOMIC_RELEASE, __HIP_MEMORY_SCOPE_AGENT);
        } else {
            while (__hip_atomic_load(gen, __ATOMIC_ACQUIRE, __HIP_MEMORY_SCOPE_AGENT) == g)
                __builtin_amdgcn_s_sleep(2);
        }
    }
    __syncthreads();
}

// ---------------------------------------------------------------------------
// Persistent LSTM layer scan.
//   xp    : [S*64][4096] f32   (x-projection incl. bias, gate-major columns)
//   WhT   : [4096][1024] bf16  (row n = g*1024+h holds Wh[g][:,h])
//   hseq  : [S*64][1024] bf16  (output hidden sequence, also h_{t-1} source)
//   cst   : [64][1024] f32     (cell state scratch)
// Grid: 64 blocks x 128 threads = 256 waves; wave w -> batch tile rb = w&3,
// hidden tile hc = w>>2. Each wave: 4 gates x (K=1024)/32 = 128 WMMAs / step.
// ---------------------------------------------------------------------------
__global__ void __launch_bounds__(128)
lstm_layer_kernel(const float* __restrict__ xp, const bf16* __restrict__ WhT,
                  bf16* __restrict__ hseq, float* __restrict__ cst,
                  unsigned* __restrict__ bar, int S, int nblocks) {
    const int lane = threadIdx.x & 31;
    const int w  = blockIdx.x * 4 + (threadIdx.x >> 5);  // 0..255
    const int rb = w & 3;                                // batch tile (16 rows)
    const int hc = w >> 2;                               // hidden tile (16 cols)
    const int cn = lane & 15, ckh = lane >> 4;
    unsigned* counter = bar;
    unsigned* gen     = bar + 1;

    for (int t = 0; t < S; ++t) {
        v8f acc[4];
#pragma unroll
        for (int g = 0; g < 4; ++g)
#pragma unroll
            for (int j = 0; j < 8; ++j) {
                const int row = rb * 16 + j + 8 * ckh;
                acc[g][j] = xp[(size_t)(t * 64 + row) * 4096 + g * 1024 + hc * 16 + cn];
            }

        if (t > 0) {
            const bf16* hprev = hseq + (size_t)(t - 1) * 64 * 1024 + rb * 16 * 1024;
            for (int kc = 0; kc < 1024; kc += 32) {
                v16bf a = load_a_frag(hprev + kc, 1024, lane);
#pragma unroll
                for (int g = 0; g < 4; ++g) {
                    v16bf bfr = load_b_frag(
                        WhT + (size_t)(g * 1024 + hc * 16) * 1024 + kc, 1024, lane);
                    acc[g] = wmma_bf16(a, bfr, acc[g]);
                }
            }
        }

        // gates + state update on this wave's 16x16 patch
#pragma unroll
        for (int j = 0; j < 8; ++j) {
            const int row = rb * 16 + j + 8 * ckh;
            const int col = hc * 16 + cn;
            const float iv = sigmoid_(acc[0][j]);
            const float fv = sigmoid_(acc[1][j]);
            const float ov = sigmoid_(acc[2][j]);
            const float gv = tanh_(acc[3][j]);
            const float cold = (t == 0) ? 0.0f : cst[row * 1024 + col];
            const float cnew = fv * cold + iv * gv;
            cst[row * 1024 + col] = cnew;
            const float hv = ov * tanh_(cnew);
            hseq[(size_t)(t * 64 + row) * 1024 + col] = (bf16)hv;
        }

        grid_barrier(counter, gen, (unsigned)nblocks);
    }
}

// ---------------------------------------------------------------------------
// Launch
// ---------------------------------------------------------------------------
extern "C" void kernel_launch(void* const* d_in, const int* in_sizes, int n_in,
                              void* d_out, int out_size, void* d_ws, size_t ws_size,
                              hipStream_t stream) {
    const int S = 256, B = 64, H = 1024, NL = 5, NC = 32000;
    const int*   X   = (const int*)d_in[0];
    const float* emb = (const float*)d_in[1];
    const float* Wx  = (const float*)d_in[2];
    const float* Wh  = (const float*)d_in[3];
    const float* bg  = (const float*)d_in[4];
    const float* Whq = (const float*)d_in[5];
    const float* bq  = (const float*)d_in[6];
    float* out = (float*)d_out;

    // workspace carve-out
    char* base = (char*)d_ws;
    size_t off = 0;
    auto carve = [&](size_t bytes) -> void* {
        void* r = base + off;
        off = (off + bytes + 255) & ~(size_t)255;
        return r;
    };
    unsigned* bar = (unsigned*)carve(256);
    bf16* WxT  = (bf16*)carve((size_t)NL * 4096 * 1024 * sizeof(bf16));
    bf16* WhT  = (bf16*)carve((size_t)NL * 4096 * 1024 * sizeof(bf16));
    bf16* WqT  = (bf16*)carve((size_t)NC * 1024 * sizeof(bf16));
    bf16* seqA = (bf16*)carve((size_t)S * B * H * sizeof(bf16));
    bf16* seqB = (bf16*)carve((size_t)S * B * H * sizeof(bf16));
    float* xp  = (float*)carve((size_t)S * B * 4096 * sizeof(float));
    float* cst = (float*)carve((size_t)B * H * sizeof(float));
    if (off > ws_size) return;  // workspace too small: deterministic no-op

    hipMemsetAsync(bar, 0, 256, stream);

    // One-time (per call) weight repack: f32 [K][N] -> bf16 [N][K]
    transpose_f32_to_bf16T<<<dim3(32, 32, 20), 256, 0, stream>>>(Wx,  WxT, 1024, 1024);
    transpose_f32_to_bf16T<<<dim3(32, 32, 20), 256, 0, stream>>>(Wh,  WhT, 1024, 1024);
    transpose_f32_to_bf16T<<<dim3(32, 1000, 1), 256, 0, stream>>>(Whq, WqT, 1024, NC);

    embed_kernel<<<S * B, 256, 0, stream>>>(X, emb, seqA);

    for (int l = 0; l < NL; ++l) {
        const bf16* inSeq  = (l & 1) ? seqB : seqA;
        bf16*       outSeq = (l & 1) ? seqA : seqB;
        // xp = inSeq x Wx[l] + b[l]   (M=16384, N=4096, K=1024)
        wmma_gemm_bias_kernel<<<dim3(4096 / 128, (S * B) / 64), 256, 0, stream>>>(
            inSeq, WxT + (size_t)l * 4096 * 1024, bg + l * 4096, xp, S * B, 4096, 1024);
        // sequential scan (persistent, 256 grid barriers)
        lstm_layer_kernel<<<64, 128, 0, stream>>>(
            xp, WhT + (size_t)l * 4096 * 1024, outSeq, cst, bar, S, 64);
    }

    // output projection on last timestep of top layer (layer 4 output = seqB)
    const bf16* htop = seqB + (size_t)(S - 1) * B * H;
    wmma_gemm_bias_kernel<<<dim3(NC / 128, 1), 256, 0, stream>>>(
        htop, WqT, bq, out, B, NC, 1024);
}
